// GroupingTransformer_17051020165803
// MI455X (gfx1250) — matrix operand order repaired and checked by reading, not compiled, not run on hardware
//
#include <hip/hip_runtime.h>

// ---------------------------------------------------------------------------
// Types for CDNA5 WMMA
// ---------------------------------------------------------------------------
typedef __attribute__((ext_vector_type(16))) _Float16 v16h;
typedef __attribute__((ext_vector_type(8)))  float    v8f;
typedef __attribute__((ext_vector_type(2)))  _Float16 h2;

#define DEVI __device__ __forceinline__

DEVI int lane_id() { return threadIdx.x & 31; }
DEVI int wave_id() { return threadIdx.x >> 5; }

DEVI v8f wmma_f16(v16h a, v16h b, v8f c) {
  // D = A(16x32 f16) * B(32x16 f16) + C(16x16 f32)
  return __builtin_amdgcn_wmma_f32_16x16x32_f16(
      /*neg_a=*/false, a, /*neg_b=*/false, b,
      /*c_mod=*/(short)0, c, /*reuse_a=*/false, /*reuse_b=*/false);
}

// ---------------------------------------------------------------------------
// Fragment loaders (contiguous k order, k = 16*(lane/16) + element, for A and B)
// ---------------------------------------------------------------------------
// A-matrix 16x32 f16: lanes 0-15 -> M=lane, K=0..15;
//                     lanes 16-31 -> M=lane-16, K=16..31
DEVI v16h frag_a(const _Float16* A, int lda, int m0, int k0) {
  int l = lane_id();
  const _Float16* row = A + (m0 + (l & 15)) * lda;
  int hsel = (l >> 4) * 16;
  v16h r;
#pragma unroll
  for (int v = 0; v < 8; ++v) {
    int k = k0 + hsel + v * 2;
    h2 p = *(const h2*)(row + k);
    r[2 * v] = p[0];
    r[2 * v + 1] = p[1];
  }
  return r;
}

// B-matrix 32x16 f16 where B = S^T and S (NxK) is row-major in LDS.
// B[k][n] = S[n][k]  -> contiguous pair reads.
DEVI v16h frag_b_tr(const _Float16* S, int lds_, int n0, int k0) {
  int l = lane_id();
  const _Float16* row = S + (n0 + (l & 15)) * lds_;
  int kh = (l >> 4) * 16;
  v16h r;
#pragma unroll
  for (int v = 0; v < 8; ++v) {
    int k = k0 + kh + 2 * v;
    h2 p = *(const h2*)(row + k);
    r[2 * v] = p[0];
    r[2 * v + 1] = p[1];
  }
  return r;
}

// B-matrix 32x16 f16 from row-major KxN LDS (strided element reads).
DEVI v16h frag_b_rm(const _Float16* Bm, int ldb, int k0, int n0) {
  int l = lane_id();
  int n = n0 + (l & 15);
  int kh = (l >> 4) * 16;
  v16h r;
#pragma unroll
  for (int v = 0; v < 8; ++v) {
    int k = k0 + kh + 2 * v;
    r[2 * v]     = Bm[k * ldb + n];
    r[2 * v + 1] = Bm[(k + 1) * ldb + n];
  }
  return r;
}

// B-matrix from fragment-packed weights: one aligned 32B load per lane.
DEVI v16h frag_b_packed(const _Float16* P, int wNT, int kt, int nt) {
  return *(const v16h*)(P + ((size_t)(kt * wNT + nt) * 32 + lane_id()) * 16);
}

// D (16x16 f32) store into f16 LDS row-major, with optional bias/resid/relu.
// Rows >= valid_rows are stored as 0 (padding hygiene).
DEVI void store_tile_f16(_Float16* O, int ldo, int mt, int nt, v8f d,
                         int valid_rows, const float* bias,
                         const _Float16* resid, bool relu) {
  int l = lane_id();
  int n = nt * 16 + (l & 15);
  int mb = mt * 16 + (l >> 4) * 8;
  float bv = bias ? bias[n] : 0.f;
#pragma unroll
  for (int v = 0; v < 8; ++v) {
    int m = mb + v;
    float val = d[v] + bv;
    if (resid) val += (float)resid[m * ldo + n];
    if (relu) val = fmaxf(val, 0.f);
    O[m * ldo + n] = (m < valid_rows) ? (_Float16)val : (_Float16)0.f;
  }
}

// ---------------------------------------------------------------------------
// Generic tiled GEMMs (4 waves per block, one 16x16 output tile per step)
// ---------------------------------------------------------------------------
template <typename Epi>
DEVI void gemm_w(const _Float16* A, int lda, const _Float16* Wp, int wNT,
                 int MT, int NT, int KT, int kt0, int nt0, Epi epi) {
  for (int t = wave_id(); t < MT * NT; t += 4) {
    int mt = t / NT, nt = t % NT;
    v8f acc = {};
    for (int kt = 0; kt < KT; ++kt) {
      v16h a = frag_a(A, lda, mt * 16, kt * 32);
      v16h b = frag_b_packed(Wp, wNT, kt0 + kt, nt0 + nt);
      acc = wmma_f16(a, b, acc);
    }
    epi(mt, nt, acc);
  }
}

template <typename Epi>
DEVI void gemm_tr(const _Float16* A, int lda, const _Float16* S, int lds_,
                  int MT, int NT, int KT, Epi epi) {
  for (int t = wave_id(); t < MT * NT; t += 4) {
    int mt = t / NT, nt = t % NT;
    v8f acc = {};
    for (int kt = 0; kt < KT; ++kt) {
      v16h a = frag_a(A, lda, mt * 16, kt * 32);
      v16h b = frag_b_tr(S, lds_, nt * 16, kt * 32);
      acc = wmma_f16(a, b, acc);
    }
    epi(mt, nt, acc);
  }
}

template <typename Epi>
DEVI void gemm_rm(const _Float16* A, int lda, const _Float16* Bm, int ldb,
                  int MT, int NT, int KT, Epi epi) {
  for (int t = wave_id(); t < MT * NT; t += 4) {
    int mt = t / NT, nt = t % NT;
    v8f acc = {};
    for (int kt = 0; kt < KT; ++kt) {
      v16h a = frag_a(A, lda, mt * 16, kt * 32);
      v16h b = frag_b_rm(Bm, ldb, kt * 32, nt * 16);
      acc = wmma_f16(a, b, acc);
    }
    epi(mt, nt, acc);
  }
}

// ---------------------------------------------------------------------------
// LayerNorm: 2 threads per row, pair-reduction via wave32 lane swap.
// ---------------------------------------------------------------------------
DEVI void layernorm_rows(_Float16* X, const float* g, const float* be) {
  int r = threadIdx.x >> 1;
  int c0 = (threadIdx.x & 1) * 64;
  _Float16* row = X + r * 128;
  float m = 0.f;
  for (int c = 0; c < 64; ++c) m += (float)row[c0 + c];
  m += __shfl_xor(m, 1, 32);
  m *= (1.f / 128.f);
  float var = 0.f;
  for (int c = 0; c < 64; ++c) { float d = (float)row[c0 + c] - m; var += d * d; }
  var += __shfl_xor(var, 1, 32);
  var *= (1.f / 128.f);
  float inv = rsqrtf(var + 1e-5f);
  for (int c = 0; c < 64; ++c) {
    int cc = c0 + c;
    row[cc] = (_Float16)(((float)row[cc] - m) * inv * g[cc] + be[cc]);
  }
}

// ---------------------------------------------------------------------------
// Multi-head attention core (4 heads, dh=32). Writes concat(o_h) to attnout.
// Xq: 64x128 f16; Xkv: (mtkv*16)x128 f16; Pw: 4 packed 128x128 projections.
// ---------------------------------------------------------------------------
DEVI void attention_block(const _Float16* Xq, const _Float16* Xkv, int mtkv,
                          int kvalid, int qvalid, bool causal,
                          const _Float16* Pw, const float* bias,
                          _Float16* bufQ, _Float16* bufK, _Float16* bufS,
                          _Float16* attnout) {
  for (int h = 0; h < 4; ++h) {
    int nt0 = h * 2;
    // Q = Xq @ Wq[:, h*32 : (h+1)*32]
    gemm_w(Xq, 128, Pw + 0 * 16384, 8, 4, 2, 4, 0, nt0,
           [&](int mt, int nt, v8f d) {
             store_tile_f16(bufQ, 32, mt, nt, d, 64, bias + 0 * 128 + h * 32,
                            nullptr, false);
           });
    // K = Xkv @ Wk[:, ...]
    gemm_w(Xkv, 128, Pw + 1 * 16384, 8, mtkv, 2, 4, 0, nt0,
           [&](int mt, int nt, v8f d) {
             store_tile_f16(bufK, 32, mt, nt, d, kvalid,
                            bias + 1 * 128 + h * 32, nullptr, false);
           });
    __syncthreads();
    // S = (Q @ K^T) / sqrt(32)
    gemm_tr(bufQ, 32, bufK, 32, 4, 4, 1, [&](int mt, int nt, v8f d) {
      int l = lane_id();
      int col = nt * 16 + (l & 15);
      int mb = mt * 16 + (l >> 4) * 8;
#pragma unroll
      for (int v = 0; v < 8; ++v)
        bufS[(mb + v) * 64 + col] = (_Float16)(d[v] * 0.1767766953f);
    });
    __syncthreads();
    // V = Xkv @ Wv (overwrites bufQ; Q is dead) -- concurrent with softmax.
    gemm_w(Xkv, 128, Pw + 2 * 16384, 8, mtkv, 2, 4, 0, nt0,
           [&](int mt, int nt, v8f d) {
             store_tile_f16(bufQ, 32, mt, nt, d, kvalid,
                            bias + 2 * 128 + h * 32, nullptr, false);
           });
    // softmax rows of S: 2 threads per row, pair-reduce via lane swap.
    {
      int r = threadIdx.x >> 1;
      int lo = (threadIdx.x & 1) * 32, hi = lo + 32;
      _Float16* row = bufS + r * 64;
      if (r < qvalid) {
        int jmax = causal ? ((r + 1) < kvalid ? (r + 1) : kvalid) : kvalid;
        int jend = jmax < hi ? jmax : hi;
        float mx = -1e30f;
        for (int j = lo; j < jend; ++j) mx = fmaxf(mx, (float)row[j]);
        mx = fmaxf(mx, __shfl_xor(mx, 1, 32));
        float ssum = 0.f;
        for (int j = lo; j < jend; ++j) ssum += __expf((float)row[j] - mx);
        ssum += __shfl_xor(ssum, 1, 32);
        float inv = 1.f / ssum;
        for (int j = lo; j < hi; ++j)
          row[j] = (j < jmax) ? (_Float16)(__expf((float)row[j] - mx) * inv)
                              : (_Float16)0.f;
      } else {
        for (int j = lo; j < hi; ++j) row[j] = (_Float16)0.f;
      }
    }
    __syncthreads();
    // o_h = softmax(S) @ V -> attnout[:, h*32 : (h+1)*32]
    gemm_rm(bufS, 64, bufQ, 32, 4, 2, 2, [&](int mt, int nt, v8f d) {
      store_tile_f16(attnout + h * 32, 128, mt, nt, d, 64, nullptr, nullptr,
                     false);
    });
    __syncthreads();
  }
}

// FFN: X = X + relu(X @ W1 + b1) @ W2 + b2 (pre-LN stored back into X).
// K=512 chunked by 128; second GEMM accumulates in registers (8 tiles/wave).
DEVI void ffn_block(_Float16* X, _Float16* T, const _Float16* Pw1,
                    const _Float16* Pw2, const float* b1_, const float* b2_,
                    int valid) {
  v8f zero = {};
  v8f acc[8];
#pragma unroll
  for (int i = 0; i < 8; ++i) acc[i] = zero;
  for (int chunk = 0; chunk < 4; ++chunk) {
    __syncthreads();
    gemm_w(X, 128, Pw1, 32, 4, 8, 4, 0, chunk * 8, [&](int mt, int nt, v8f d) {
      store_tile_f16(T, 128, mt, nt, d, 64, b1_ + chunk * 128, nullptr, true);
    });
    __syncthreads();
    int w = wave_id();
#pragma unroll
    for (int ti = 0; ti < 8; ++ti) {
      int t = w + ti * 4;
      int mt = t >> 3, nt = t & 7;
#pragma unroll
      for (int kt = 0; kt < 4; ++kt) {
        v16h a = frag_a(T, 128, mt * 16, kt * 32);
        v16h b = frag_b_packed(Pw2, 8, chunk * 4 + kt, nt);
        acc[ti] = wmma_f16(a, b, acc[ti]);
      }
    }
  }
  __syncthreads();
  int w = wave_id();
#pragma unroll
  for (int ti = 0; ti < 8; ++ti) {
    int t = w + ti * 4;
    int mt = t >> 3, nt = t & 7;
    store_tile_f16(X, 128, mt, nt, acc[ti], valid, b2_, X, false);
  }
}

// ---------------------------------------------------------------------------
// Weight packing: f32 KxN -> fragment-ordered f16 tiles (32x16 per tile)
// ---------------------------------------------------------------------------
__global__ __launch_bounds__(32) void pack_w_kernel(const float* W, _Float16* P,
                                                    int K, int N) {
  int NT = N >> 4;
  int tile = blockIdx.x;
  int kt = tile / NT, nt = tile % NT;
  int l = threadIdx.x;
  int n = nt * 16 + (l & 15);
  int kb = kt * 32 + (l >> 4) * 16;
  _Float16* out = P + ((size_t)tile * 32 + l) * 16;
#pragma unroll
  for (int i = 0; i < 16; ++i)
    out[i] = (_Float16)W[(size_t)(kb + i) * N + n];
  (void)K;
}

// ---------------------------------------------------------------------------
// Encoder: one block per (b, agent) sequence. T=50 padded to 64 rows.
// ---------------------------------------------------------------------------
__global__ __launch_bounds__(128) void enc_kernel(
    const float* __restrict__ obs, const float* __restrict__ in_b,
    const float* __restrict__ tpos, const float* __restrict__ attn_b,
    const float* __restrict__ ln_g, const float* __restrict__ ln_b,
    const float* __restrict__ b1, const float* __restrict__ b2,
    const float* __restrict__ agent_embed, const _Float16* __restrict__ Pinw,
    const _Float16* __restrict__ Pattn, const _Float16* __restrict__ Pw1,
    const _Float16* __restrict__ Pw2, _Float16* __restrict__ memf) {
  __shared__ __align__(16) _Float16 sX[64 * 128];  // activations
  __shared__ __align__(16) _Float16 sO[64 * 128];  // obs / attnout / ffn T
  __shared__ __align__(16) _Float16 sQ[64 * 32];
  __shared__ __align__(16) _Float16 sK[64 * 32];
  __shared__ __align__(16) _Float16 sS[64 * 64];

  int s = blockIdx.x;           // 0..2047
  int b = s >> 5, n = s & 31;
  int tid = threadIdx.x;

  // warm L2 for the weight streams (global_prefetch)
  if (tid < 4) {
    __builtin_prefetch(Pattn + tid * 16384, 0, 0);
    __builtin_prefetch(Pw1 + tid * 16384, 0, 0);
    __builtin_prefetch(Pw2 + tid * 16384, 0, 0);
  }

  // obs (50x64) -> f16 LDS (ld 64, rows 50..63 zero)
  for (int i = tid; i < 64 * 64; i += 128) {
    int t = i >> 6, c = i & 63;
    float v = (t < 50) ? obs[((size_t)s * 50 + t) * 64 + c] : 0.f;
    sO[i] = (_Float16)v;
  }
  __syncthreads();

  // X = obs @ in_w + in_b + temporal_pos   (K=64 -> KT=2)
  gemm_w(sO, 64, Pinw, 8, 4, 8, 2, 0, 0, [&](int mt, int nt, v8f d) {
    int l = lane_id();
    int col = nt * 16 + (l & 15);
    int mb = mt * 16 + (l >> 4) * 8;
    float bv = in_b[col];
#pragma unroll
    for (int v = 0; v < 8; ++v) {
      int m = mb + v;
      // unconditional clamped load + select: avoids exec-mask divergence
      float tp = tpos[(m < 50 ? m : 0) * 128 + col];
      float val = d[v] + bv + (m < 50 ? tp : 0.f);
      sX[m * 128 + col] = (m < 50) ? (_Float16)val : (_Float16)0.f;
    }
  });
  __syncthreads();

  for (int layer = 0; layer < 2; ++layer) {
    const _Float16* Pw = Pattn + (size_t)layer * 4 * 16384;
    const float* bw = attn_b + layer * 4 * 128;
    attention_block(sX, sX, 4, 50, 50, /*causal=*/false, Pw, bw, sQ, sK, sS, sO);
    // out-proj + residual (pre-LN into sX)
    gemm_w(sO, 128, Pw + 3 * 16384, 8, 4, 8, 4, 0, 0,
           [&](int mt, int nt, v8f d) {
             store_tile_f16(sX, 128, mt, nt, d, 50, bw + 3 * 128, sX, false);
           });
    __syncthreads();
    layernorm_rows(sX, ln_g + (layer * 2 + 0) * 128, ln_b + (layer * 2 + 0) * 128);
    __syncthreads();
    ffn_block(sX, sO, Pw1 + (size_t)layer * 65536, Pw2 + (size_t)layer * 65536,
              b1 + layer * 512, b2 + layer * 128, 50);
    __syncthreads();
    layernorm_rows(sX, ln_g + (layer * 2 + 1) * 128, ln_b + (layer * 2 + 1) * 128);
    __syncthreads();
  }

  // agent_emb = mean over T(=50) + agent_embed[n]
  {
    int c = tid;
    if (c < 128) {
      float sum = 0.f;
      for (int t = 0; t < 50; ++t) sum += (float)sX[t * 128 + c];
      float val = sum * (1.f / 50.f) + agent_embed[n * 128 + c];
      memf[((size_t)b * 32 + n) * 128 + c] = (_Float16)val;
    }
  }
}

// ---------------------------------------------------------------------------
// Decoder: one block per batch element. L=64, memory 34 rows padded to 48.
// ---------------------------------------------------------------------------
__global__ __launch_bounds__(128) void dec_kernel(
    const _Float16* __restrict__ memf, const float* __restrict__ eoe,
    const float* __restrict__ eos, const float* __restrict__ bos,
    const float* __restrict__ edge_emb, const int* __restrict__ tgt,
    const float* __restrict__ sa_b, const float* __restrict__ ca_b,
    const float* __restrict__ ln_g, const float* __restrict__ ln_b,
    const float* __restrict__ b1, const float* __restrict__ b2,
    const float* __restrict__ out_b, const _Float16* __restrict__ Psa,
    const _Float16* __restrict__ Pca, const _Float16* __restrict__ Pw1,
    const _Float16* __restrict__ Pw2, const _Float16* __restrict__ Poutw,
    float* __restrict__ out) {
  __shared__ __align__(16) _Float16 dH[64 * 128];
  __shared__ __align__(16) _Float16 dM[48 * 128];
  __shared__ __align__(16) _Float16 dO[64 * 128];  // attnout / ffn T / z
  __shared__ __align__(16) _Float16 dQ[64 * 32];
  __shared__ __align__(16) _Float16 dK[64 * 32];
  __shared__ __align__(16) _Float16 dS[64 * 64];
  __shared__ int sEdge[64];
  __shared__ unsigned sUsed[64];
  __shared__ unsigned char sEmpty[64];
  __shared__ unsigned char sCov[64];
  __shared__ int sClosed[64];

  int b = blockIdx.x;
  int tid = threadIdx.x;

  // ---- memory rows 0..31: async DMA from workspace straight into LDS ----
  // 32 rows x 128 f16 = 8192 B contiguous; 512 x b128 transfers, ASYNCcnt.
  {
    const _Float16* src = memf + (size_t)b * 32 * 128;
    unsigned lbase = (unsigned)(uintptr_t)dM;
#pragma unroll
    for (int it = 0; it < 4; ++it) {
      unsigned off = (unsigned)(tid + it * 128) * 16u;
      asm volatile("global_load_async_to_lds_b128 %0, %1, %2"
                   :
                   : "v"(lbase + off), "v"(off), "s"(src)
                   : "memory");
    }
    asm volatile("s_wait_asynccnt 0x0" ::: "memory");
  }
  // rows 32 = eoe, 33 = eos, 34..47 zero
  for (int i = 32 * 128 + tid; i < 48 * 128; i += 128) {
    int r = i >> 7, c = i & 127;
    _Float16 v;
    if (r == 32)      v = (_Float16)eoe[c];
    else if (r == 33) v = (_Float16)eos[c];
    else              v = (_Float16)0.f;
    dM[i] = v;
  }
  // sequential grouping-state scan (one thread; 64 steps, trivial)
  if (tid == 0) {
    unsigned any = 0u, in_ = 0u;
    int eidx = 0, closed = 0;
    bool empty = true;
    for (int l2 = 0; l2 < 64; ++l2) {
      sEdge[l2] = eidx;
      sUsed[l2] = any;
      sEmpty[l2] = empty ? 1 : 0;
      sCov[l2] = (any == 0xffffffffu) ? 1 : 0;
      sClosed[l2] = closed;
      int tok = tgt[b * 64 + l2];
      bool ia = tok < 32;
      bool ie = tok == 32;
      unsigned oh = ia ? (1u << tok) : 0u;
      any |= oh;
      in_ = ie ? 0u : (in_ | oh);
      eidx += ie ? 1 : 0;
      closed += ie ? 1 : 0;
      empty = ie ? true : (ia ? false : empty);
    }
    (void)in_;
  }
  __syncthreads();

  // h init: token embed (gathered from memory / bos) + edge embed + sinusoid
  for (int i = tid; i < 64 * 128; i += 128) {
    int l2 = i >> 7, c = i & 127;
    int prev = (l2 == 0) ? -1 : tgt[b * 64 + l2 - 1];
    int safe = prev < 0 ? 0 : (prev > 33 ? 33 : prev);
    float te = (prev < 0) ? bos[c] : (float)dM[safe * 128 + c];
    int ei = sEdge[l2] < 32 ? sEdge[l2] : 32;
    int ii = c >> 1;
    float freq = __expf(-logf(10000.f) * (float)(2 * ii) * (1.f / 128.f));
    float ang = (float)l2 * freq;
    float pe = (c & 1) ? __cosf(ang) : __sinf(ang);
    dH[i] = (_Float16)(te + edge_emb[ei * 128 + c] + pe);
  }
  __syncthreads();

  for (int layer = 0; layer < 2; ++layer) {
    // ---- causal self-attention ----
    const _Float16* Ps = Psa + (size_t)layer * 4 * 16384;
    const float* bs = sa_b + layer * 4 * 128;
    attention_block(dH, dH, 4, 64, 64, /*causal=*/true, Ps, bs, dQ, dK, dS, dO);
    gemm_w(dO, 128, Ps + 3 * 16384, 8, 4, 8, 4, 0, 0,
           [&](int mt, int nt, v8f d) {
             store_tile_f16(dH, 128, mt, nt, d, 64, bs + 3 * 128, dH, false);
           });
    __syncthreads();
    layernorm_rows(dH, ln_g + (layer * 3 + 0) * 128, ln_b + (layer * 3 + 0) * 128);
    __syncthreads();
    // ---- cross-attention to memory (34 valid rows, 3 row-tiles) ----
    const _Float16* Pc = Pca + (size_t)layer * 4 * 16384;
    const float* bc = ca_b + layer * 4 * 128;
    attention_block(dH, dM, 3, 34, 64, /*causal=*/false, Pc, bc, dQ, dK, dS, dO);
    gemm_w(dO, 128, Pc + 3 * 16384, 8, 4, 8, 4, 0, 0,
           [&](int mt, int nt, v8f d) {
             store_tile_f16(dH, 128, mt, nt, d, 64, bc + 3 * 128, dH, false);
           });
    __syncthreads();
    layernorm_rows(dH, ln_g + (layer * 3 + 1) * 128, ln_b + (layer * 3 + 1) * 128);
    __syncthreads();
    // ---- FFN ----
    ffn_block(dH, dO, Pw1 + (size_t)layer * 65536, Pw2 + (size_t)layer * 65536,
              b1 + layer * 512, b2 + layer * 128, 64);
    __syncthreads();
    layernorm_rows(dH, ln_g + (layer * 3 + 2) * 128, ln_b + (layer * 3 + 2) * 128);
    __syncthreads();
  }

  // z = h @ out_w + out_b  ->  dO
  gemm_w(dH, 128, Poutw, 8, 4, 8, 4, 0, 0, [&](int mt, int nt, v8f d) {
    store_tile_f16(dO, 128, mt, nt, d, 64, out_b, nullptr, false);
  });
  __syncthreads();

  // logits = z @ memory^T / sqrt(128), with legality mask fused into store
  gemm_tr(dO, 128, dM, 128, 4, 3, 4, [&](int mt, int nt, v8f d) {
    int l = lane_id();
    int col = nt * 16 + (l & 15);
    int mb = mt * 16 + (l >> 4) * 8;
    if (col < 34) {
#pragma unroll
      for (int v = 0; v < 8; ++v) {
        int row = mb + v;
        bool at_cap = sClosed[row] >= 32;
        bool empty = sEmpty[row] != 0;
        bool ok;
        if (col < 32)
          ok = (((sUsed[row] >> col) & 1u) == 0u) && !(at_cap && empty);
        else if (col == 32)
          ok = !empty && !at_cap;
        else
          ok = ((sCov[row] != 0) && empty) || at_cap;
        float val = d[v] * 0.0883883476f;  // 1/sqrt(128)
        out[((size_t)b * 64 + row) * 34 + col] = ok ? val : -__builtin_inff();
      }
    }
  });
}

// ---------------------------------------------------------------------------
// Host launcher
// ---------------------------------------------------------------------------
extern "C" void kernel_launch(void* const* d_in, const int* in_sizes, int n_in,
                              void* d_out, int out_size, void* d_ws,
                              size_t ws_size, hipStream_t stream) {
  (void)in_sizes; (void)n_in; (void)out_size; (void)ws_size;
  const float* obs        = (const float*)d_in[0];
  const float* in_w       = (const float*)d_in[1];
  const float* in_b       = (const float*)d_in[2];
  const float* tpos       = (const float*)d_in[3];
  const float* agent_emb  = (const float*)d_in[4];
  const float* eoe_e      = (const float*)d_in[5];
  const float* eos_e      = (const float*)d_in[6];
  const float* bos_e      = (const float*)d_in[7];
  const float* edge_emb   = (const float*)d_in[8];
  const float* enc_attn_w = (const float*)d_in[9];
  const float* enc_attn_b = (const float*)d_in[10];
  const float* enc_ln_g   = (const float*)d_in[11];
  const float* enc_ln_b   = (const float*)d_in[12];
  const float* enc_w1     = (const float*)d_in[13];
  const float* enc_b1     = (const float*)d_in[14];
  const float* enc_w2     = (const float*)d_in[15];
  const float* enc_b2     = (const float*)d_in[16];
  const float* dec_sa_w   = (const float*)d_in[17];
  const float* dec_sa_b   = (const float*)d_in[18];
  const float* dec_ca_w   = (const float*)d_in[19];
  const float* dec_ca_b   = (const float*)d_in[20];
  const float* dec_ln_g   = (const float*)d_in[21];
  const float* dec_ln_b   = (const float*)d_in[22];
  const float* dec_w1     = (const float*)d_in[23];
  const float* dec_b1     = (const float*)d_in[24];
  const float* dec_w2     = (const float*)d_in[25];
  const float* dec_b2     = (const float*)d_in[26];
  const float* out_w      = (const float*)d_in[27];
  const float* out_b      = (const float*)d_in[28];
  const int*   tgt        = (const int*)d_in[29];
  float* out = (float*)d_out;

  // Workspace layout (all 256B-aligned)
  _Float16* ws = (_Float16*)d_ws;
  size_t off = 0;
  auto alloc = [&](size_t halves) {
    _Float16* p = ws + off;
    off += (halves + 127) & ~(size_t)127;
    return p;
  };
  _Float16* Pinw  = alloc(64 * 128);
  _Float16* Penc  = alloc(2 * 4 * 128 * 128);
  _Float16* Pew1  = alloc((size_t)2 * 128 * 512);
  _Float16* Pew2  = alloc((size_t)2 * 512 * 128);
  _Float16* Psa   = alloc(2 * 4 * 128 * 128);
  _Float16* Pca   = alloc(2 * 4 * 128 * 128);
  _Float16* Pdw1  = alloc((size_t)2 * 128 * 512);
  _Float16* Pdw2  = alloc((size_t)2 * 512 * 128);
  _Float16* Poutw = alloc(128 * 128);
  _Float16* memf  = alloc((size_t)64 * 32 * 128);

  // ---- pack all GEMM weights into fragment-ordered f16 ----
  pack_w_kernel<<<16, 32, 0, stream>>>(in_w, Pinw, 64, 128);
  for (int i = 0; i < 8; ++i) {
    pack_w_kernel<<<32, 32, 0, stream>>>(enc_attn_w + (size_t)i * 16384,
                                         Penc + (size_t)i * 16384, 128, 128);
    pack_w_kernel<<<32, 32, 0, stream>>>(dec_sa_w + (size_t)i * 16384,
                                         Psa + (size_t)i * 16384, 128, 128);
    pack_w_kernel<<<32, 32, 0, stream>>>(dec_ca_w + (size_t)i * 16384,
                                         Pca + (size_t)i * 16384, 128, 128);
  }
  for (int l = 0; l < 2; ++l) {
    pack_w_kernel<<<128, 32, 0, stream>>>(enc_w1 + (size_t)l * 65536,
                                          Pew1 + (size_t)l * 65536, 128, 512);
    pack_w_kernel<<<128, 32, 0, stream>>>(enc_w2 + (size_t)l * 65536,
                                          Pew2 + (size_t)l * 65536, 512, 128);
    pack_w_kernel<<<128, 32, 0, stream>>>(dec_w1 + (size_t)l * 65536,
                                          Pdw1 + (size_t)l * 65536, 128, 512);
    pack_w_kernel<<<128, 32, 0, stream>>>(dec_w2 + (size_t)l * 65536,
                                          Pdw2 + (size_t)l * 65536, 512, 128);
  }
  pack_w_kernel<<<32, 32, 0, stream>>>(out_w, Poutw, 128, 128);

  // ---- temporal encoder: 2048 sequences, fully fused in LDS ----
  enc_kernel<<<2048, 128, 0, stream>>>(obs, in_b, tpos, enc_attn_b, enc_ln_g,
                                       enc_ln_b, enc_b1, enc_b2, agent_emb,
                                       Pinw, Penc, Pew1, Pew2, memf);

  // ---- decoder + logits + legality mask: one block per batch element ----
  dec_kernel<<<64, 128, 0, stream>>>(memf, eoe_e, eos_e, bos_e, edge_emb, tgt,
                                     dec_sa_b, dec_ca_b, dec_ln_g, dec_ln_b,
                                     dec_b1, dec_b2, out_b, Psa, Pca, Pdw1,
                                     Pdw2, Poutw, out);
}
